// GCN_47837345743091
// MI455X (gfx1250) — matrix operand order repaired and checked by reading, not compiled, not run on hardware
//
#include <hip/hip_runtime.h>
#include <hip/hip_bf16.h>

typedef __attribute__((ext_vector_type(16))) _Float16 v16h;
typedef __attribute__((ext_vector_type(8)))  _Float16 v8h;
typedef __attribute__((ext_vector_type(8)))  float    v8f;

#define GCN_N   50000
#define GCN_E   800000
#define GCN_F   128
#define HID     64
#define EPSV    1e-5f

// ---------------------------------------------------------------- utilities
__global__ void k_fill_f32(float* __restrict__ p, float v, int n) {
    int i = blockIdx.x * blockDim.x + threadIdx.x;
    if (i < n) p[i] = v;
}

__global__ void k_cvt_f16(const float* __restrict__ src, _Float16* __restrict__ dst, int n) {
    int i = blockIdx.x * blockDim.x + threadIdx.x;
    if (i < n) dst[i] = (_Float16)src[i];
}

// ---------------------------------------------------------------- degree / norm
__global__ void k_deg_count(const int* __restrict__ dst, float* __restrict__ deg, int e) {
    int i = blockIdx.x * blockDim.x + threadIdx.x;
    if (i < e) atomicAdd(&deg[dst[i]], 1.0f);
}

__global__ void k_dinv(const float* __restrict__ deg, float* __restrict__ dinv, int n) {
    int i = blockIdx.x * blockDim.x + threadIdx.x;
    if (i < n) dinv[i] = rsqrtf(deg[i]);   // deg >= 1 always (self loop)
}

// ---------------------------------------------------------------- WMMA GEMM
// C[M x 64] = A_f16[M x K] @ B_f16[K x 64], f32 accumulate.
// One wave per 16x16 output tile. M % 16 == 0, K % 32 == 0.
__global__ void k_gemm_wmma(const _Float16* __restrict__ A,
                            const _Float16* __restrict__ B,
                            float* __restrict__ C, int M, int K) {
    int wave = (int)((blockIdx.x * blockDim.x + threadIdx.x) >> 5);
    int lane = threadIdx.x & 31;
    int nTilesTotal = (M >> 4) * (HID >> 4);
    if (wave >= nTilesTotal) return;

    const int m0 = (wave >> 2) << 4;   // M tile base
    const int n0 = (wave & 3)  << 4;   // N tile base (HID/16 == 4)

    const int hiA  = lane >> 4;        // 0: lanes 0-15, 1: lanes 16-31
    const int aRow = m0 + (lane & 15);

    v8f c = {};
    for (int kk = 0; kk < K; kk += 32) {
        // A 16-bit 16x32 layout: halfs 0-7 = K[k0..k0+7], halfs 8-15 = K[k0+16..k0+23],
        // k0 = kk + 8*hiA  (per ISA 7.12.2 table)
        const int k0 = kk + (hiA << 3);
        v8h a_lo = *(const v8h*)(A + (size_t)aRow * K + k0);
        v8h a_hi = *(const v8h*)(A + (size_t)aRow * K + k0 + 16);
        v16h a = __builtin_shufflevector(a_lo, a_hi,
                                         0,1,2,3,4,5,6,7,8,9,10,11,12,13,14,15);
        // B 16-bit 32x16: lane L holds column N = n0 + L%16, K in same order as A
        const int bCol = n0 + (lane & 15);
        v16h b;
#pragma unroll
        for (int s = 0; s < 16; ++s)
            b[s] = B[(size_t)(k0 + (s < 8 ? s : s + 8)) * HID + bCol];

        c = __builtin_amdgcn_wmma_f32_16x16x32_f16(
                /*neg_a=*/false, a, /*neg_b=*/false, b,
                /*c_mod=*/(short)0, c, /*reuse_a=*/false, /*reuse_b=*/false);
    }

    // C/D layout: VGPR r -> (M = m0 + r + 8*hiA, N = n0 + lane&15)
    const int n  = n0 + (lane & 15);
    const int mb = m0 + (hiA << 3);
#pragma unroll
    for (int r = 0; r < 8; ++r)
        C[(size_t)(mb + r) * HID + n] = c[r];
}

// ---------------------------------------------------------------- edge scatter
// item < e: edge (src[item] -> dst[item]); item >= e: self loop (item - e).
// One wave per item, 2 features per lane, f32 atomics (resolve in L2).
__global__ void k_scatter(const float* __restrict__ T,
                          const int* __restrict__ src, const int* __restrict__ dst,
                          const float* __restrict__ dinv,
                          float* __restrict__ agg, int e, int n) {
    long long gt = (long long)blockIdx.x * blockDim.x + threadIdx.x;
    int item = (int)(gt >> 5);
    int lane = (int)(gt & 31);
    if (item >= e + n) return;

    int s, d;
    if (item < e) { s = src[item]; d = dst[item]; }
    else          { s = d = item - e; }

    float nm = dinv[s] * dinv[d];
    float2 v = *(const float2*)(T + (size_t)s * HID + lane * 2);
    atomicAdd(&agg[(size_t)d * HID + lane * 2],     v.x * nm);
    atomicAdd(&agg[(size_t)d * HID + lane * 2 + 1], v.y * nm);
}

// ---------------------------------------------------------------- batchnorm
__global__ void k_bn_stats(const float* __restrict__ h, float* __restrict__ stats, int n) {
    int f  = threadIdx.x & 63;
    int rpb = blockDim.x >> 6;                        // rows per block per pass
    int r0 = blockIdx.x * rpb + (threadIdx.x >> 6);
    int stride = gridDim.x * rpb;
    float s = 0.f, s2 = 0.f;
    for (int r = r0; r < n; r += stride) {
        float v = h[(size_t)r * HID + f];
        s += v; s2 += v * v;
    }
    atomicAdd(&stats[f],      s);
    atomicAdd(&stats[64 + f], s2);
}

// normalize + affine + ReLU, emit f16 for the next WMMA layer
__global__ void k_bn_apply(const float* __restrict__ h, const float* __restrict__ stats,
                           const float* __restrict__ g, const float* __restrict__ be,
                           _Float16* __restrict__ out, int n) {
    int i = blockIdx.x * blockDim.x + threadIdx.x;
    if (i >= n * HID) return;
    int f = i & 63;
    float inv_n = 1.0f / (float)n;
    float mu  = stats[f] * inv_n;
    float var = stats[64 + f] * inv_n - mu * mu;
    float y = (h[i] - mu) * rsqrtf(var + EPSV) * g[f] + be[f];
    out[i] = (_Float16)fmaxf(y, 0.f);
}

// ---------------------------------------------------------------- final relu + fc
__global__ void k_final(const float* __restrict__ agg, const float* __restrict__ b2,
                        const float* __restrict__ fcW, const float* __restrict__ fcb,
                        float* __restrict__ out, int n) {
    int i = blockIdx.x * blockDim.x + threadIdx.x;
    if (i >= n) return;
    float acc = fcb[0];
#pragma unroll
    for (int f = 0; f < HID; ++f) {
        float v = agg[(size_t)i * HID + f] + b2[f];
        acc += fmaxf(v, 0.f) * fcW[f];
    }
    out[i] = acc;
}

// ---------------------------------------------------------------- launcher
static inline int cdiv(long long a, int b) { return (int)((a + b - 1) / b); }

extern "C" void kernel_launch(void* const* d_in, const int* in_sizes, int n_in,
                              void* d_out, int out_size, void* d_ws, size_t ws_size,
                              hipStream_t stream) {
    (void)in_sizes; (void)n_in; (void)out_size; (void)ws_size;

    const float* x   = (const float*)d_in[0];
    const int*   ei  = (const int*)d_in[1];
    const float* W0  = (const float*)d_in[2];
    // d_in[3] = b0 : folds out of BN exactly (shifts mean only)
    const float* W1  = (const float*)d_in[4];
    // d_in[5] = b1 : folds out of BN
    const float* W2  = (const float*)d_in[6];
    const float* b2  = (const float*)d_in[7];
    const float* g0  = (const float*)d_in[8];
    const float* be0 = (const float*)d_in[9];
    const float* g1  = (const float*)d_in[10];
    const float* be1 = (const float*)d_in[11];
    const float* fcW = (const float*)d_in[12];
    const float* fcb = (const float*)d_in[13];
    float* out = (float*)d_out;

    const int* e_src = ei;
    const int* e_dst = ei + GCN_E;

    // --- workspace carve (256B aligned) ---
    char* w = (char*)d_ws;
    auto carve = [&](size_t bytes) -> char* {
        char* p = w; w += (bytes + 255) & ~(size_t)255; return p;
    };
    float*     deg   = (float*)carve((size_t)GCN_N * 4);
    float*     dinv  = (float*)carve((size_t)GCN_N * 4);
    float*     stats = (float*)carve(128 * 4);
    _Float16*  Xh    = (_Float16*)carve((size_t)GCN_N * GCN_F * 2);
    _Float16*  W0h   = (_Float16*)carve((size_t)GCN_F * HID * 2);
    _Float16*  W1h   = (_Float16*)carve((size_t)HID * HID * 2);
    _Float16*  W2h   = (_Float16*)carve((size_t)HID * HID * 2);
    _Float16*  Hh    = (_Float16*)carve((size_t)GCN_N * HID * 2);
    float*     T     = (float*)carve((size_t)GCN_N * HID * 4);
    float*     agg   = (float*)carve((size_t)GCN_N * HID * 4);

    const int B = 256;
    const int NH = GCN_N * HID;                       // 3.2M
    const int gemmWaves = (GCN_N / 16) * (HID / 16);  // 12500
    const int gemmBlocks = cdiv((long long)gemmWaves * 32, B);
    const int scatBlocks = cdiv((long long)(GCN_E + GCN_N) * 32, B);

    // degrees / symmetric norm
    k_fill_f32 <<<cdiv(GCN_N, B), B, 0, stream>>>(deg, 1.0f, GCN_N); // self loop
    k_deg_count<<<cdiv(GCN_E, B), B, 0, stream>>>(e_dst, deg, GCN_E);
    k_dinv     <<<cdiv(GCN_N, B), B, 0, stream>>>(deg, dinv, GCN_N);

    // f16 conversions
    k_cvt_f16<<<cdiv((long long)GCN_N * GCN_F, B), B, 0, stream>>>(x,  Xh,  GCN_N * GCN_F);
    k_cvt_f16<<<cdiv(GCN_F * HID, B), B, 0, stream>>>(W0, W0h, GCN_F * HID);
    k_cvt_f16<<<cdiv(HID * HID, B), B, 0, stream>>>(W1, W1h, HID * HID);
    k_cvt_f16<<<cdiv(HID * HID, B), B, 0, stream>>>(W2, W2h, HID * HID);

    // ---- layer 0: conv(x) -> BN -> ReLU ----
    k_gemm_wmma<<<gemmBlocks, B, 0, stream>>>(Xh, W0h, T, GCN_N, GCN_F);
    k_fill_f32 <<<cdiv(NH, B), B, 0, stream>>>(agg, 0.0f, NH);
    k_scatter  <<<scatBlocks, B, 0, stream>>>(T, e_src, e_dst, dinv, agg, GCN_E, GCN_N);
    k_fill_f32 <<<1, 128, 0, stream>>>(stats, 0.0f, 128);
    k_bn_stats <<<128, B, 0, stream>>>(agg, stats, GCN_N);
    k_bn_apply <<<cdiv(NH, B), B, 0, stream>>>(agg, stats, g0, be0, Hh, GCN_N);

    // ---- layer 1: conv(h) -> BN -> ReLU ----
    k_gemm_wmma<<<gemmBlocks, B, 0, stream>>>(Hh, W1h, T, GCN_N, HID);
    k_fill_f32 <<<cdiv(NH, B), B, 0, stream>>>(agg, 0.0f, NH);
    k_scatter  <<<scatBlocks, B, 0, stream>>>(T, e_src, e_dst, dinv, agg, GCN_E, GCN_N);
    k_fill_f32 <<<1, 128, 0, stream>>>(stats, 0.0f, 128);
    k_bn_stats <<<128, B, 0, stream>>>(agg, stats, GCN_N);
    k_bn_apply <<<cdiv(NH, B), B, 0, stream>>>(agg, stats, g1, be1, Hh, GCN_N);

    // ---- layer 2: conv(h) -> ReLU -> fc ----
    k_gemm_wmma<<<gemmBlocks, B, 0, stream>>>(Hh, W2h, T, GCN_N, HID);
    k_fill_f32 <<<cdiv(NH, B), B, 0, stream>>>(agg, 0.0f, NH);
    k_scatter  <<<scatBlocks, B, 0, stream>>>(T, e_src, e_dst, dinv, agg, GCN_E, GCN_N);
    k_final    <<<cdiv(GCN_N, B), B, 0, stream>>>(agg, b2, fcW, fcb, out, GCN_N);
}
